// Network_89953795048154
// MI455X (gfx1250) — compile-verified, matrix operand order checked by reading, not run on hardware
//
#include <hip/hip_runtime.h>
#include <stdint.h>

typedef __attribute__((ext_vector_type(16))) _Float16 v16h;
typedef __attribute__((ext_vector_type(8)))  float    v8f;

union B16Frag {
    uint32_t u[8];
    uint4    q[2];
    v16h     h;
};

static __device__ inline float silu_f(float x) {
    return x / (1.0f + __expf(-x));
}

static __device__ inline v16h cvt_frag(float4 a0, float4 a1, float4 b0, float4 b1) {
    v16h f;
    f[0]  = (_Float16)a0.x; f[1]  = (_Float16)a0.y;
    f[2]  = (_Float16)a0.z; f[3]  = (_Float16)a0.w;
    f[4]  = (_Float16)a1.x; f[5]  = (_Float16)a1.y;
    f[6]  = (_Float16)a1.z; f[7]  = (_Float16)a1.w;
    f[8]  = (_Float16)b0.x; f[9]  = (_Float16)b0.y;
    f[10] = (_Float16)b0.z; f[11] = (_Float16)b0.w;
    f[12] = (_Float16)b1.x; f[13] = (_Float16)b1.y;
    f[14] = (_Float16)b1.z; f[15] = (_Float16)b1.w;
    return f;
}

// ---------------------------------------------------------------------------
// Pre-swizzle W1f (64x256) and W2f (16x64) into f16 wave32 B-matrix fragment
// layout.  Storage: wsB[(tile*32 + lane)*8 + v]  (per-lane contiguous dwords,
// so the hot loop loads each fragment as two b128s).
//   layer1: tile = s*4 + j  (s = k-step 0..7, j = n-tile 0..3)   -> 8192 dwords
//   layer2: tile = 32 + s2  (s2 = 0..1)                          ->  512 dwords
// Within a fragment: half h = lane>>4, N = 16*j + (lane&15),
//   k0 = (v<4) ? (2v + 8h) : (16 + 2(v-4) + 8h), K = 32*step + k0, pair (K,K+1).
// B[k][n] = W.T[k][n] = W[n][k].
// ---------------------------------------------------------------------------
__global__ void prep_weights(const float* __restrict__ W1f,
                             const float* __restrict__ W2f,
                             uint32_t* __restrict__ wsB) {
    int e = blockIdx.x * blockDim.x + threadIdx.x;
    if (e < 8192) {
        int tile = e >> 8, lane = (e >> 3) & 31, v = e & 7;
        int s = tile >> 2, j = tile & 3;
        int h = lane >> 4, n = j * 16 + (lane & 15);
        int k0 = (v < 4) ? (2 * v + 8 * h) : (16 + 2 * (v - 4) + 8 * h);
        int k = 32 * s + k0;
        union { _Float16 hv[2]; uint32_t u; } p;
        p.hv[0] = (_Float16)W1f[n * 256 + k];
        p.hv[1] = (_Float16)W1f[n * 256 + k + 1];
        wsB[e] = p.u;
    } else if (e < 8704) {
        int e2 = e - 8192;
        int s2 = e2 >> 8, lane = (e2 >> 3) & 31, v = e2 & 7;
        int h = lane >> 4, n = lane & 15;
        int k0 = (v < 4) ? (2 * v + 8 * h) : (16 + 2 * (v - 4) + 8 * h);
        int k = 32 * s2 + k0;
        union { _Float16 hv[2]; uint32_t u; } p;
        p.hv[0] = (_Float16)W2f[n * 64 + k];
        p.hv[1] = (_Float16)W2f[n * 64 + k + 1];
        wsB[e] = p.u;
    }
}

// ---------------------------------------------------------------------------
// Copies + constant fills.  d_out float layout (element offsets):
//   [0,512)            energy           (copy)
//   [512,154112)       forces           (copy)
//   [154112,158720)    stress           (copy)
//   [158720,159232)    energy_uncert    = 0.6
//   [159232,312832)    force_uncert     (main kernel writes this)
//   [312832,317440)    stress_uncert    = 0.00625
// ---------------------------------------------------------------------------
__global__ void copy_fill_kernel(const float* __restrict__ energy,
                                 const float* __restrict__ forces,
                                 const float* __restrict__ stress,
                                 float* __restrict__ out) {
    int i = blockIdx.x * blockDim.x + threadIdx.x;
    if (i < 512)          out[i] = energy[i];
    else if (i < 154112)  out[i] = forces[i - 512];
    else if (i < 158720)  out[i] = stress[i - 154112];
    else if (i < 159232)  out[i] = 0.6f;
    else if (i < 163840)  out[312832 + (i - 159232)] = 0.00625f;
}

// ---------------------------------------------------------------------------
// F-head MLP via WMMA.  One wave per 16-atom tile (3200 tiles).
// ---------------------------------------------------------------------------
__global__ __launch_bounds__(128) void fu_mlp_kernel(
    const float* __restrict__ node_feats,      // [51200, 384]
    const uint32_t* __restrict__ wsB,          // pre-swizzled f16 weights
    const float* __restrict__ b1f,             // [64]
    const float* __restrict__ b2f,             // [16]
    const float* __restrict__ W3f,             // [1,16]
    const float* __restrict__ b3f,             // [1]
    float* __restrict__ out_fu) {              // [51200, 3]
    __shared__ float h1s[4][16][64];           // per-wave layer1 activations

    const int lane = threadIdx.x & 31;
    const int wave = threadIdx.x >> 5;
    const int h    = lane >> 4;                // lane half
    const int ln   = lane & 15;
    const int tile = blockIdx.x * 4 + wave;
    const int atom = tile * 16 + ln;           // A-layout: lane -> row M
    const float* rp = node_feats + (size_t)atom * 384;

    // ---- layer 1: [16x256] @ [256x64], 4 N-tiles x 8 K-steps --------------
    v8f acc[4];
#pragma unroll
    for (int j = 0; j < 4; ++j) acc[j] = (v8f)(0.0f);

#pragma unroll
    for (int s = 0; s < 8; ++s) {
        // nf column runs for this lane: [cA, cA+8) and [cB, cB+8)
        int cA = 32 * s + 8 * h;
        int cB = cA + 16;
        // nf col c -> node_feats col: c (c<128) else c+128 (last-128 block)
        int mA = cA + ((cA >= 128) ? 128 : 0);
        int mB = cB + ((cB >= 128) ? 128 : 0);
        float4 a0 = *(const float4*)(rp + mA);
        float4 a1 = *(const float4*)(rp + mA + 4);
        float4 c0 = *(const float4*)(rp + mB);
        float4 c1 = *(const float4*)(rp + mB + 4);
        v16h afrag = cvt_frag(a0, a1, c0, c1);
#pragma unroll
        for (int j = 0; j < 4; ++j) {
            B16Frag bf;
            const uint4* bp = (const uint4*)(wsB + (size_t)((s * 4 + j) * 32 + lane) * 8);
            bf.q[0] = bp[0];
            bf.q[1] = bp[1];
            acc[j] = __builtin_amdgcn_wmma_f32_16x16x32_f16(
                false, afrag, false, bf.h, (short)0, acc[j], false, false);
        }
    }

    // bias + SiLU, then write to LDS row-major for the layout transpose
    // (C/D layout: elem v of acc[j] is h1[M = v + 8h][N = 16j + ln])
#pragma unroll
    for (int j = 0; j < 4; ++j) {
        float bias = b1f[j * 16 + ln];
#pragma unroll
        for (int e = 0; e < 8; ++e) {
            h1s[wave][e + 8 * h][j * 16 + ln] = silu_f(acc[j][e] + bias);
        }
    }
    __syncthreads();

    // ---- layer 2: [16x64] @ [64x16], 2 K-steps ----------------------------
    v8f acc2 = (v8f)(0.0f);
#pragma unroll
    for (int s2 = 0; s2 < 2; ++s2) {
        const float* hp = &h1s[wave][ln][0];
        int c0i = 32 * s2 + 8 * h;
        float4 a0 = *(const float4*)(hp + c0i);
        float4 a1 = *(const float4*)(hp + c0i + 4);
        float4 d0 = *(const float4*)(hp + c0i + 16);
        float4 d1 = *(const float4*)(hp + c0i + 20);
        v16h afrag = cvt_frag(a0, a1, d0, d1);
        B16Frag bf;
        const uint4* bp = (const uint4*)(wsB + 8192 + (size_t)(s2 * 32 + lane) * 8);
        bf.q[0] = bp[0];
        bf.q[1] = bp[1];
        acc2 = __builtin_amdgcn_wmma_f32_16x16x32_f16(
            false, afrag, false, bf.h, (short)0, acc2, false, false);
    }

    // ---- layer 3 (16 -> 1) + exp: per-M cross-lane reduction --------------
    float bias2 = b2f[ln];
    float w3    = W3f[ln];
    float b3v   = b3f[0];
    float fu[8];
#pragma unroll
    for (int e = 0; e < 8; ++e) {
        float p = silu_f(acc2[e] + bias2) * w3;   // h2[M][N=ln] * W3[ln]
        p += __shfl_xor(p, 1, 32);                // reduce within each 16-lane
        p += __shfl_xor(p, 2, 32);                // half (lanes 0-15: M=e,
        p += __shfl_xor(p, 4, 32);                //  lanes 16-31: M=e+8)
        p += __shfl_xor(p, 8, 32);
        fu[e] = 0.1f * __expf(p + b3v);
    }

    if (ln == 0) {                                 // lanes 0 and 16 store
#pragma unroll
        for (int e = 0; e < 8; ++e) {
            int a = tile * 16 + e + 8 * h;
            float v = fu[e];
            out_fu[(size_t)a * 3 + 0] = v;
            out_fu[(size_t)a * 3 + 1] = v;
            out_fu[(size_t)a * 3 + 2] = v;
        }
    }
}

extern "C" void kernel_launch(void* const* d_in, const int* in_sizes, int n_in,
                              void* d_out, int out_size, void* d_ws, size_t ws_size,
                              hipStream_t stream) {
    const float* node_feats = (const float*)d_in[0];
    const float* energy     = (const float*)d_in[1];
    const float* forces     = (const float*)d_in[2];
    const float* stress     = (const float*)d_in[3];
    const float* W1f        = (const float*)d_in[10];
    const float* b1f        = (const float*)d_in[11];
    const float* W2f        = (const float*)d_in[12];
    const float* b2f        = (const float*)d_in[13];
    const float* W3f        = (const float*)d_in[14];
    const float* b3f        = (const float*)d_in[15];
    float*    out = (float*)d_out;
    uint32_t* wsB = (uint32_t*)d_ws;

    // 1) swizzle weights into WMMA B-fragment layout (8704 dwords, ~34 KB ws)
    prep_weights<<<34, 256, 0, stream>>>(W1f, W2f, wsB);
    // 2) copies + constant regions (independent of 1)
    copy_fill_kernel<<<640, 256, 0, stream>>>(energy, forces, stress, out);
    // 3) WMMA MLP -> force_uncert (after 1, same stream => ordered)
    fu_mlp_kernel<<<800, 128, 0, stream>>>(node_feats, wsB, b1f, b2f, W3f, b3f,
                                           out + 159232);
}
